// NNConvModel_84567906058925
// MI455X (gfx1250) — matrix operand order repaired and checked by reading, not compiled, not run on hardware
//
#include <hip/hip_runtime.h>
#include <hip/hip_bf16.h>

typedef _Float16 f16;
typedef __attribute__((ext_vector_type(16))) _Float16 v16h;
typedef __attribute__((ext_vector_type(8)))  float    v8f;

#define LEAKK 0.1f
static __device__ __forceinline__ float leaky(float v) { return fmaxf(v, LEAKK * v); }

// Position in the A-fragment-contiguous ("swizzled") storage for logical K in [0,32).
// g=0 reads pos[0..15] -> K = {0..7,16..23}; g=1 reads pos[16..31] -> K = {8..15,24..31}.
// (ISA 7.12.2, 16-bit A-matrix 16x32 layout)
static __device__ __forceinline__ int posOfK(int K) {
  if (K < 8)  return K;
  if (K < 16) return K + 8;   // stored at 16..23
  if (K < 24) return K - 8;   // stored at 8..15
  return K;                   // 24..31
}

static __device__ __forceinline__ v8f wmma_f16(v16h a, v16h b, v8f c) {
  return __builtin_amdgcn_wmma_f32_16x16x32_f16(false, a, false, b, (short)0, c, false, false);
}

// ---------------------------------------------------------------------------
// BatchNorm statistics: deterministic two-pass column reduction (cols <= 32)
// ---------------------------------------------------------------------------
__global__ __launch_bounds__(256) void bn_partial_kernel(const float* __restrict__ src,
                                                         int rows, int cols,
                                                         float* __restrict__ partial) {
  __shared__ float red[512];
  int tid = threadIdx.x;
  int col = tid & 31, rs = tid >> 5;               // 8 row-slots x 32 cols
  float s = 0.f, q = 0.f;
  if (col < cols) {
    for (int r = blockIdx.x * 8 + rs; r < rows; r += gridDim.x * 8) {
      float v = src[(size_t)r * cols + col];
      s += v; q += v * v;
    }
  }
  red[tid] = s; red[256 + tid] = q;
  __syncthreads();
  for (int st = 4; st > 0; st >>= 1) {
    if (rs < st) {
      red[tid]       += red[tid + st * 32];
      red[256 + tid] += red[256 + tid + st * 32];
    }
    __syncthreads();
  }
  if (tid < 32) {
    partial[blockIdx.x * 64 + tid]      = red[tid];
    partial[blockIdx.x * 64 + 32 + tid] = red[256 + tid];
  }
}

__global__ void bn_finalize_kernel(const float* __restrict__ partial, int nblocks,
                                   int rows, int cols,
                                   const float* __restrict__ gamma,
                                   const float* __restrict__ beta,
                                   float* __restrict__ ss) {
  int t = threadIdx.x;
  if (t >= 32) return;
  float s = 0.f, q = 0.f;
  for (int b = 0; b < nblocks; b++) { s += partial[b * 64 + t]; q += partial[b * 64 + 32 + t]; }
  float mean = s / (float)rows;
  float var  = q / (float)rows - mean * mean;       // biased (jnp.var default)
  float g  = (t < cols) ? gamma[t] : 1.f;
  float bb = (t < cols) ? beta[t]  : 0.f;
  float scale = g * rsqrtf(var + 1e-5f);
  ss[t]      = scale;
  ss[32 + t] = bb - mean * scale;
}

// h[n, 0..31] = BN(x)[n, c] for c<16 else 0   (fp32)
__global__ __launch_bounds__(256) void bn_apply_nodes_kernel(const float* __restrict__ x,
                                                             const float* __restrict__ ss,
                                                             float* __restrict__ h, int n) {
  int idx = blockIdx.x * blockDim.x + threadIdx.x;
  if (idx >= n * 32) return;
  int c = idx & 31, r = idx >> 5;
  float v = 0.f;
  if (c < 16) v = x[(size_t)r * 16 + c] * ss[c] + ss[32 + c];
  h[idx] = v;
}

// eswz[e, p] = f16(BN(e)[e, KofPos(p)])  (A-fragment swizzled, padded K 19..31 = 0)
__global__ __launch_bounds__(256) void bn_apply_edges_kernel(const float* __restrict__ e,
                                                             const float* __restrict__ ss,
                                                             f16* __restrict__ eswz, int ne) {
  int idx = blockIdx.x * blockDim.x + threadIdx.x;
  if (idx >= ne * 32) return;
  int p = idx & 31, r = idx >> 5;
  int K = (p < 8) ? p : (p < 16) ? p + 8 : (p < 24) ? p - 8 : p;  // inverse of posOfK
  float v = 0.f;
  if (K < 19) v = e[(size_t)r * 19 + K] * ss[K] + ss[32 + K];
  eswz[idx] = (f16)v;
}

// ---------------------------------------------------------------------------
// Pack a [K x N] fp32 weight matrix into per-lane-contiguous f16 B-fragments:
//   out[((kc*ntiles + t)*32 + lane)*16 + j] = W[kc*32 + 8*(lane/16) + j + (j>=8 ? 8 : 0)][t*16 + lane%16]
// ---------------------------------------------------------------------------
__global__ __launch_bounds__(256) void pack_b_kernel(const float* __restrict__ W,
                                                     f16* __restrict__ out,
                                                     int K, int N, int kchunks, int ntiles) {
  int idx = blockIdx.x * blockDim.x + threadIdx.x;
  int total = kchunks * ntiles * 512;
  if (idx >= total) return;
  int j  = idx & 15;
  int l  = (idx >> 4) & 31;
  int t  = (idx >> 9) % ntiles;
  int kc = idx / (512 * ntiles);
  int g = l >> 4;
  int n = t * 16 + (l & 15);
  int k = kc * 32 + 8 * g + j + ((j >= 8) ? 8 : 0);
  float v = (k < K && n < N) ? W[(size_t)k * N + n] : 0.f;
  out[idx] = (f16)v;
}

// ---------------------------------------------------------------------------
// NNConv edge kernel (fused):  per 16-edge tile (one wave):
//   w1 = leaky(e_n @ W1 + b1)                  [16 x din]    (1-2 WMMA)
//   for i in 0..din-1: both 16-wide N-halves of w2 = leaky(w1 @ W2 + b2):
//       msg[:, oh*16:] += h[src, i] * w2tile   (2 WMMA per i, paired B loads)
//   atomicAdd msg into agg[dst]
// ---------------------------------------------------------------------------
__global__ __launch_bounds__(256) void edge_mp_kernel(
    const f16* __restrict__ eswz, const int* __restrict__ ei,
    const float* __restrict__ h,
    const f16* __restrict__ W1p, const float* __restrict__ b1,
    const f16* __restrict__ W2p, const float* __restrict__ b2,
    float* __restrict__ agg, int din, int nE) {
  __shared__ f16   w1s[8][16 * 32];   // per-wave w1 tile, A-swizzled
  __shared__ float hs[8][16 * 32];    // per-wave gathered h[src] tile (fp32)
  int wave = threadIdx.x >> 5, lane = threadIdx.x & 31;
  int ebase = (blockIdx.x * 8 + wave) * 16;
  if (ebase >= nE) return;                      // wave-uniform
  int m = lane & 15, g = lane >> 4;

  // gather h[src] rows into LDS (halves split columns by parity)
  int srcm = ei[ebase + m];
  for (int c = g; c < din; c += 2) hs[wave][m * 32 + c] = h[(size_t)srcm * 32 + c];

  if (din == 16) {  // zero-pad w1 K=16..31 storage positions (8..15 and 24..31)
#pragma unroll
    for (int q = 0; q < 8; q++) {
      int z = lane * 8 + q;
      int row = z >> 4, pp = z & 15;
      int pos = (pp < 8) ? pp + 8 : pp + 16;
      w1s[wave][row * 32 + pos] = (f16)0.f;
    }
  }

  // A fragment of normalized edge features (already swizzled in memory)
  v16h ae = *(const v16h*)(eswz + (size_t)(ebase + m) * 32 + g * 16);
  // warm the cache for the next tile's edge features (global_prefetch_b8)
  __builtin_prefetch(eswz + (size_t)(ebase + 16 + m) * 32 + g * 16, 0, 3);

  // w1 = leaky(e_n @ W1 + b1), spilled to LDS in A-swizzled layout
  int ntiles1 = din >> 4;
  for (int t = 0; t < ntiles1; t++) {
    v16h bf = *(const v16h*)(W1p + ((size_t)t * 32 + lane) * 16);
    v8f c = {};
    c = wmma_f16(ae, bf, c);
    float bias = b1[t * 16 + m];
    int p = posOfK(t * 16 + m);                 // column (= next-GEMM K) -> storage pos
#pragma unroll
    for (int r = 0; r < 8; r++) {
      float v = leaky(c[r] + bias);
      w1s[wave][(r + g * 8) * 32 + p] = (f16)v;
    }
  }
  v16h aw = *(const v16h*)(&w1s[wave][m * 32 + g * 16]);

  float macc[16];
#pragma unroll
  for (int k = 0; k < 16; k++) macc[k] = 0.f;

  // main loop: i indexes the h[src] feature; N-tiles 2i (o=0..15) and 2i+1 (o=16..31)
  for (int i = 0; i < din; i++) {
    float hv[8];
#pragma unroll
    for (int r = 0; r < 8; r++) hv[r] = hs[wave][(r + g * 8) * 32 + i];

    const f16* f0 = W2p + ((size_t)(2 * i) * 32 + lane) * 16;
    v16h bf0 = *(const v16h*)f0;
    v16h bf1 = *(const v16h*)(f0 + 512);                 // N-tile 2i+1
    __builtin_prefetch(f0 + 1024, 0, 3);                 // next i, first half
    __builtin_prefetch(f0 + 1536, 0, 3);                 // next i, second half

    v8f c0 = {}, c1 = {};
    c0 = wmma_f16(aw, bf0, c0);
    c1 = wmma_f16(aw, bf1, c1);

    float bias0 = b2[(2 * i) * 16 + m];
    float bias1 = b2[(2 * i + 1) * 16 + m];
#pragma unroll
    for (int r = 0; r < 8; r++) {
      macc[r]     += hv[r] * leaky(c0[r] + bias0);
      macc[8 + r] += hv[r] * leaky(c1[r] + bias1);
    }
  }

  // scatter-add messages:  element (edge = ebase+r+8g, o = oh*16 + m)
#pragma unroll
  for (int r = 0; r < 8; r++) {
    int e = ebase + r + g * 8;
    int d = ei[nE + e];
    atomicAdd(&agg[(size_t)d * 32 + m],       macc[r]);
    atomicAdd(&agg[(size_t)d * 32 + 16 + m],  macc[8 + r]);
  }
}

// h_out = h_in @ root + agg + bias          (one WMMA per 16-node x 16-col tile)
__global__ __launch_bounds__(256) void node_update_kernel(
    const float* __restrict__ hin, const f16* __restrict__ rootp,
    const float* __restrict__ bias, const float* __restrict__ agg,
    float* __restrict__ hout, int din, int nN) {
  int wave = threadIdx.x >> 5, lane = threadIdx.x & 31;
  int nbase = (blockIdx.x * 8 + wave) * 16;
  if (nbase >= nN) return;
  int m = lane & 15, g = lane >> 4;
  v16h a;
#pragma unroll
  for (int j = 0; j < 16; j++) {
    int K = g * 8 + j + ((j >= 8) ? 8 : 0);
    float v = (K < din) ? hin[(size_t)(nbase + m) * 32 + K] : 0.f;
    a[j] = (f16)v;
  }
#pragma unroll
  for (int t = 0; t < 2; t++) {
    v16h bf = *(const v16h*)(rootp + ((size_t)t * 32 + lane) * 16);
    v8f c = {};
    c = wmma_f16(a, bf, c);
    float bb = bias[t * 16 + m];
#pragma unroll
    for (int r = 0; r < 8; r++) {
      size_t row = (size_t)(nbase + r + g * 8);
      hout[row * 32 + t * 16 + m] = c[r] + bb + agg[row * 32 + t * 16 + m];
    }
  }
}

// ---------------------------------------------------------------------------
// Edge prediction MLP: [h_src | h_dst | e_n] (83) -> 64 -> 32 -> 16 -> 8 -> 2
// First three layers on WMMA (K padded to 32-chunks), 16->8->2 tail in VALU.
// ---------------------------------------------------------------------------
__global__ __launch_bounds__(256) void pred_kernel(
    const float* __restrict__ h, const f16* __restrict__ eswz,
    const int* __restrict__ ei,
    const f16* __restrict__ W0p, const float* __restrict__ b0,   // 83(->96) x 64
    const f16* __restrict__ W1p, const float* __restrict__ b1,   // 64 x 32
    const f16* __restrict__ W2p, const float* __restrict__ b2,   // 32 x 16
    const float* __restrict__ W3, const float* __restrict__ b3,  // 16 x 8
    const float* __restrict__ W4, const float* __restrict__ b4,  // 8 x 2
    float* __restrict__ out, int nE) {
  __shared__ f16 zs[8][16 * 96];
  int wave = threadIdx.x >> 5, lane = threadIdx.x & 31;
  int ebase = (blockIdx.x * 8 + wave) * 16;
  if (ebase >= nE) return;
  int m = lane & 15, g = lane >> 4;
  f16* Z = zs[wave];
  int s = ei[ebase + m];
  int d = ei[nE + ebase + m];

  // build z0 (f16, A-swizzled per 32-wide K-chunk), cols 83..95 zero
  for (int q = 0; q < 48; q++) {
    int c = g * 48 + q;
    float v = 0.f;
    if (c < 32)      v = h[(size_t)s * 32 + c];
    else if (c < 64) v = h[(size_t)d * 32 + (c - 32)];
    else if (c < 83) v = (float)eswz[(size_t)(ebase + m) * 32 + posOfK(c - 64)];
    int kc = c >> 5, K = c & 31;
    Z[m * 96 + kc * 32 + posOfK(K)] = (f16)v;
  }

  // layer 1: 83 -> 64
  v16h a0 = *(const v16h*)(Z + m * 96 +  0 + g * 16);
  v16h a1 = *(const v16h*)(Z + m * 96 + 32 + g * 16);
  v16h a2 = *(const v16h*)(Z + m * 96 + 64 + g * 16);
#pragma unroll
  for (int t = 0; t < 4; t++) {
    v8f c = {};
    c = wmma_f16(a0, *(const v16h*)(W0p + ((size_t)(0 * 4 + t) * 32 + lane) * 16), c);
    c = wmma_f16(a1, *(const v16h*)(W0p + ((size_t)(1 * 4 + t) * 32 + lane) * 16), c);
    c = wmma_f16(a2, *(const v16h*)(W0p + ((size_t)(2 * 4 + t) * 32 + lane) * 16), c);
    float bb = b0[t * 16 + m];
    int col = t * 16 + m;
    int off = (col >> 5) * 32 + posOfK(col & 31);
#pragma unroll
    for (int r = 0; r < 8; r++) Z[(r + g * 8) * 96 + off] = (f16)leaky(c[r] + bb);
  }

  // layer 2: 64 -> 32
  v16h c0 = *(const v16h*)(Z + m * 96 +  0 + g * 16);
  v16h c1 = *(const v16h*)(Z + m * 96 + 32 + g * 16);
#pragma unroll
  for (int t = 0; t < 2; t++) {
    v8f c = {};
    c = wmma_f16(c0, *(const v16h*)(W1p + ((size_t)(0 * 2 + t) * 32 + lane) * 16), c);
    c = wmma_f16(c1, *(const v16h*)(W1p + ((size_t)(1 * 2 + t) * 32 + lane) * 16), c);
    float bb = b1[t * 16 + m];
    int off = posOfK(t * 16 + m);
#pragma unroll
    for (int r = 0; r < 8; r++) Z[(r + g * 8) * 96 + off] = (f16)leaky(c[r] + bb);
  }

  // layer 3: 32 -> 16
  v16h a3 = *(const v16h*)(Z + m * 96 + g * 16);
  {
    v8f c = {};
    c = wmma_f16(a3, *(const v16h*)(W2p + (size_t)lane * 16), c);
    float bb = b2[m];
    float* Zf = (float*)Z;                       // reuse LDS as 16x16 fp32
#pragma unroll
    for (int r = 0; r < 8; r++) Zf[(r + g * 8) * 16 + m] = leaky(c[r] + bb);
  }

  // tail: 16 -> 8 -> 2 in VALU (lane owns edge m; halves duplicate, g==0 stores)
  const float* Zf = (const float*)Z;
  float row[16];
#pragma unroll
  for (int n = 0; n < 16; n++) row[n] = Zf[m * 16 + n];
  float v4[8];
#pragma unroll
  for (int o = 0; o < 8; o++) {
    float acc = b3[o];
#pragma unroll
    for (int n = 0; n < 16; n++) acc += row[n] * W3[n * 8 + o];
    v4[o] = leaky(acc);
  }
  if (g == 0) {
    float o0 = b4[0], o1 = b4[1];
#pragma unroll
    for (int o = 0; o < 8; o++) { o0 += v4[o] * W4[o * 2]; o1 += v4[o] * W4[o * 2 + 1]; }
    out[(size_t)(ebase + m) * 2]     = o0;
    out[(size_t)(ebase + m) * 2 + 1] = o1;
  }
}

// ---------------------------------------------------------------------------
extern "C" void kernel_launch(void* const* d_in, const int* in_sizes, int n_in,
                              void* d_out, int out_size, void* d_ws, size_t ws_size,
                              hipStream_t stream) {
  const int N = 65536, E = 262144;
  // input order (JAX pytree flatten, dict keys sorted):
  const float* x  = (const float*)d_in[0];
  const int*   ei = (const int*)d_in[1];
  const float* e  = (const float*)d_in[2];
  // d_in[3] = xbatch (unused)
  const float* bnE_g = (const float*)d_in[4];
  const float* bnE_b = (const float*)d_in[5];
  const float* bnN_g = (const float*)d_in[6];
  const float* bnN_b = (const float*)d_in[7];
  // mp layers: {bias, mlp[(W,b),(W,b)], root}
  const float* mpBias[3] = { (const float*)d_in[8],  (const float*)d_in[14], (const float*)d_in[20] };
  const float* mpW1[3]   = { (const float*)d_in[9],  (const float*)d_in[15], (const float*)d_in[21] };
  const float* mpB1[3]   = { (const float*)d_in[10], (const float*)d_in[16], (const float*)d_in[22] };
  const float* mpW2[3]   = { (const float*)d_in[11], (const float*)d_in[17], (const float*)d_in[23] };
  const float* mpB2[3]   = { (const float*)d_in[12], (const float*)d_in[18], (const float*)d_in[24] };
  const float* mpRoot[3] = { (const float*)d_in[13], (const float*)d_in[19], (const float*)d_in[25] };
  const float* pW[5] = { (const float*)d_in[26], (const float*)d_in[28], (const float*)d_in[30],
                         (const float*)d_in[32], (const float*)d_in[34] };
  const float* pB[5] = { (const float*)d_in[27], (const float*)d_in[29], (const float*)d_in[31],
                         (const float*)d_in[33], (const float*)d_in[35] };
  float* out = (float*)d_out;

  // ---- workspace layout ----
  float* agg = (float*)d_ws;                       // N*32 f32
  float* hA  = agg + (size_t)N * 32;               // N*32
  float* hB  = hA  + (size_t)N * 32;               // N*32
  f16*   eswz = (f16*)(hB + (size_t)N * 32);       // E*32 f16
  f16*   wp   = eswz + (size_t)E * 32;
  f16* mp0W1p = wp; wp += 1 * 512;
  f16* mp0W2p = wp; wp += 32 * 512;
  f16* mp0Rp  = wp; wp += 2 * 512;
  f16* mp1W1p = wp; wp += 2 * 512;
  f16* mp1W2p = wp; wp += 64 * 512;
  f16* mp1Rp  = wp; wp += 2 * 512;
  f16* mp2W1p = wp; wp += 2 * 512;
  f16* mp2W2p = wp; wp += 64 * 512;
  f16* mp2Rp  = wp; wp += 2 * 512;
  f16* pW0p   = wp; wp += 12 * 512;                // kc=3, nt=4
  f16* pW1p   = wp; wp += 4 * 512;                 // kc=2, nt=2
  f16* pW2p   = wp; wp += 1 * 512;                 // kc=1, nt=1
  float* partN = (float*)wp;                       // 128*64
  float* partE = partN + 128 * 64;                 // 256*64
  float* ssN   = partE + 256 * 64;                 // 64
  float* ssE   = ssN + 64;                         // 64
  f16* mpW1p_[3] = { mp0W1p, mp1W1p, mp2W1p };
  f16* mpW2p_[3] = { mp0W2p, mp1W2p, mp2W2p };
  f16* mpRp_[3]  = { mp0Rp,  mp1Rp,  mp2Rp };

  // ---- BatchNorm (training-mode batch stats) ----
  bn_partial_kernel<<<128, 256, 0, stream>>>(x, N, 16, partN);
  bn_finalize_kernel<<<1, 32, 0, stream>>>(partN, 128, N, 16, bnN_g, bnN_b, ssN);
  bn_partial_kernel<<<256, 256, 0, stream>>>(e, E, 19, partE);
  bn_finalize_kernel<<<1, 32, 0, stream>>>(partE, 256, E, 19, bnE_g, bnE_b, ssE);
  bn_apply_nodes_kernel<<<(N * 32) / 256, 256, 0, stream>>>(x, ssN, hA, N);
  bn_apply_edges_kernel<<<(E * 32) / 256, 256, 0, stream>>>(e, ssE, eswz, E);

  // ---- pack weight matrices into B fragments (f16) ----
  const int din_[3] = { 16, 32, 32 };
  for (int l = 0; l < 3; l++) {
    int din = din_[l];
    int nt1 = din / 16, nt2 = din * 2;
    pack_b_kernel<<<(nt1 * 512 + 255) / 256, 256, 0, stream>>>(mpW1[l], mpW1p_[l], 19, din, 1, nt1);
    pack_b_kernel<<<(nt2 * 512 + 255) / 256, 256, 0, stream>>>(mpW2[l], mpW2p_[l], din, din * 32, 1, nt2);
    pack_b_kernel<<<(2 * 512 + 255) / 256, 256, 0, stream>>>(mpRoot[l], mpRp_[l], din, 32, 1, 2);
  }
  pack_b_kernel<<<(12 * 512 + 255) / 256, 256, 0, stream>>>(pW[0], pW0p, 83, 64, 3, 4);
  pack_b_kernel<<<(4 * 512 + 255) / 256, 256, 0, stream>>>(pW[1], pW1p, 64, 32, 2, 2);
  pack_b_kernel<<<(1 * 512 + 255) / 256, 256, 0, stream>>>(pW[2], pW2p, 32, 16, 1, 1);

  // ---- 3 NNConv message-passing layers ----
  float* hc = hA;
  float* hn = hB;
  for (int l = 0; l < 3; l++) {
    hipMemsetAsync(agg, 0, (size_t)N * 32 * sizeof(float), stream);
    edge_mp_kernel<<<E / 16 / 8, 256, 0, stream>>>(eswz, ei, hc, mpW1p_[l], mpB1[l],
                                                   mpW2p_[l], mpB2[l], agg, din_[l], E);
    node_update_kernel<<<N / 16 / 8, 256, 0, stream>>>(hc, mpRp_[l], mpBias[l], agg, hn, din_[l], N);
    float* t = hc; hc = hn; hn = t;
  }

  // ---- edge prediction MLP ----
  pred_kernel<<<E / 16 / 8, 256, 0, stream>>>(hc, eswz, ei,
                                              pW0p, pB[0], pW1p, pB[1], pW2p, pB[2],
                                              pW[3], pB[3], pW[4], pB[4], out, E);
  (void)in_sizes; (void)n_in; (void)out_size; (void)ws_size;
}
